// MultiHeadAttention_81810537054518
// MI455X (gfx1250) — matrix operand order repaired and checked by reading, not compiled, not run on hardware
//
#include <hip/hip_runtime.h>

typedef _Float16 h16;
typedef __attribute__((ext_vector_type(16))) _Float16 v16h;
typedef __attribute__((ext_vector_type(8)))  _Float16 v8h;
typedef __attribute__((ext_vector_type(4)))  _Float16 v4h;
typedef __attribute__((ext_vector_type(8)))  float    v8f;

#define BB   4
#define SS   2048
#define DIN  1024
#define DM   1024
#define HH   16
#define DH   64
#define MTOT (BB*SS)   // 8192

#if defined(__gfx1250__) && __has_builtin(__builtin_amdgcn_global_load_async_to_lds_b128) && __has_builtin(__builtin_amdgcn_s_wait_asynccnt)
#define USE_ASYNC 1
typedef __attribute__((__vector_size__(16))) int i128v;
typedef __attribute__((address_space(1))) i128v* as1_i128p;
typedef __attribute__((address_space(3))) i128v* as3_i128p;
// copy 16 bytes global -> LDS through the async engine (ASYNCcnt-tracked)
#define CP16(g, l)                                                            \
  __builtin_amdgcn_global_load_async_to_lds_b128(                             \
      (as1_i128p)(g), (as3_i128p)(l), 0, 0)
#define WAIT_ASYNC() __builtin_amdgcn_s_wait_asynccnt(0)
#else
#define USE_ASYNC 0
#endif

#if defined(__gfx1250__) && __has_builtin(__builtin_amdgcn_exp2f)
#define EXP2F(x) __builtin_amdgcn_exp2f(x)
#else
#define EXP2F(x) exp2f(x)
#endif

#if defined(__gfx1250__) && __has_builtin(__builtin_amdgcn_permlane16)
#define USE_PERMLANE 1
__device__ __forceinline__ float pl16(float v, unsigned s0, unsigned s1) {
  unsigned u = __float_as_uint(v);
  unsigned r = __builtin_amdgcn_permlane16(u, u, s0, s1, false, false);
  return __uint_as_float(r);
}
#else
#define USE_PERMLANE 0
#endif

// butterfly reductions over the 16-lane half-wave group
__device__ __forceinline__ float redmax16(float v) {
#if USE_PERMLANE
  v = fmaxf(v, pl16(v, 0x67452301u, 0xEFCDAB89u));  // xor 1
  v = fmaxf(v, pl16(v, 0x54761032u, 0xDCFE98BAu));  // xor 2
  v = fmaxf(v, pl16(v, 0x32107654u, 0xBA98FEDCu));  // xor 4
  v = fmaxf(v, pl16(v, 0xFEDCBA98u, 0x76543210u));  // xor 8
#else
  for (int off = 1; off < 16; off <<= 1) v = fmaxf(v, __shfl_xor(v, off, 32));
#endif
  return v;
}
__device__ __forceinline__ float redsum16(float v) {
#if USE_PERMLANE
  v += pl16(v, 0x67452301u, 0xEFCDAB89u);
  v += pl16(v, 0x54761032u, 0xDCFE98BAu);
  v += pl16(v, 0x32107654u, 0xBA98FEDCu);
  v += pl16(v, 0xFEDCBA98u, 0x76543210u);
#else
  for (int off = 1; off < 16; off <<= 1) v += __shfl_xor(v, off, 32);
#endif
  return v;
}

__device__ __forceinline__ v8f wmma16(v16h a, v16h b, v8f c) {
  return __builtin_amdgcn_wmma_f32_16x16x32_f16(false, a, false, b, (short)0, c, false, false);
}

// ---------------------------------------------------------------- converts
__global__ __launch_bounds__(256) void cvt_f32_f16(const float* __restrict__ src,
                                                   h16* __restrict__ dst, int n) {
  int i = (blockIdx.x * 256 + threadIdx.x) * 4;
  if (i + 3 < n) {
    float4 v = *(const float4*)(src + i);
    v4h o;
    o[0] = (h16)v.x; o[1] = (h16)v.y; o[2] = (h16)v.z; o[3] = (h16)v.w;
    *(v4h*)(dst + i) = o;
  }
}

// W[Kd x Nd] f32 -> Wt[Nd x Kd] f16 (scaled)
__global__ __launch_bounds__(256) void cvt_tr(const float* __restrict__ W,
                                              h16* __restrict__ Wt,
                                              int Kd, int Nd, float scale) {
  __shared__ float tile[32][33];
  const int k0 = blockIdx.x * 32, n0 = blockIdx.y * 32;
  const int t = threadIdx.x;
  const int r = t >> 3, c4 = (t & 7) * 4;
  float4 v = *(const float4*)(W + (size_t)(k0 + r) * Nd + n0 + c4);
  tile[r][c4 + 0] = v.x; tile[r][c4 + 1] = v.y;
  tile[r][c4 + 2] = v.z; tile[r][c4 + 3] = v.w;
  __syncthreads();
  v4h o;
  o[0] = (h16)(tile[c4 + 0][r] * scale);
  o[1] = (h16)(tile[c4 + 1][r] * scale);
  o[2] = (h16)(tile[c4 + 2][r] * scale);
  o[3] = (h16)(tile[c4 + 3][r] * scale);
  *(v4h*)(Wt + (size_t)(n0 + r) * Kd + k0 + c4) = o;
}

// ---------------------------------------------------------------- GEMM
// C[M x N] = A[M x K] @ Wt[N x K]^T + bias*bscale.  A, Wt f16 row-major.
// 128x128 block, 8 waves in 4x2 grid, each 32x64 via 2x4 WMMA tiles, K-step 32.
// OM: 0 = f32 [MxN], 1 = f16 [MxN], 2 = f16 head-split transposed [B,H,DH,S]
template <int OM>
__global__ __launch_bounds__(256)
void gemm_wmma(const h16* __restrict__ A, const h16* __restrict__ Wt,
               const float* __restrict__ bias, float bscale,
               void* __restrict__ Cout, int M, int N, int K) {
  __shared__ h16 Asl[2][128 * 40];   // [row][k] k-chunk 32, pad to 40
  __shared__ h16 Btl[2][128 * 40];   // [n][k]

  const int tid  = threadIdx.x;
  const int wave = tid >> 5;
  const int lane = tid & 31;
  const int hl   = lane >> 4;
  const int l16  = lane & 15;
  const int wm   = wave >> 1;     // 0..3
  const int wn   = wave & 1;      // 0..1
  const int m0   = blockIdx.y * 128;
  const int n0   = blockIdx.x * 128;

  // per-thread staging slots: row = tid>>1 (0..127), 32B chunk = (tid&1)*16 halves
  const int srow = tid >> 1;
  const int scol = (tid & 1) * 16;
  const h16* gA = A  + (size_t)(m0 + srow) * K + scol;
  const h16* gB = Wt + (size_t)(n0 + srow) * K + scol;

  v8f acc[2][4];
#pragma unroll
  for (int mt = 0; mt < 2; ++mt)
#pragma unroll
    for (int nt = 0; nt < 4; ++nt)
#pragma unroll
      for (int r = 0; r < 8; ++r) acc[mt][nt][r] = 0.f;

  const int iters = K >> 5;

#if USE_ASYNC
  {
    h16* la = &Asl[0][srow * 40 + scol];
    h16* lb = &Btl[0][srow * 40 + scol];
    CP16(gA, la); CP16(gA + 8, la + 8);
    CP16(gB, lb); CP16(gB + 8, lb + 8);
  }
#endif

  for (int i = 0; i < iters; ++i) {
    const int cur = i & 1;
#if USE_ASYNC
    WAIT_ASYNC();
    __syncthreads();
    if (i + 1 < iters) {  // prefetch next K-tile into the other buffer
      const int k1 = (i + 1) * 32;
      h16* la = &Asl[1 - cur][srow * 40 + scol];
      h16* lb = &Btl[1 - cur][srow * 40 + scol];
      CP16(gA + k1, la); CP16(gA + k1 + 8, la + 8);
      CP16(gB + k1, lb); CP16(gB + k1 + 8, lb + 8);
    }
#else
    const int k0 = i * 32;
    __syncthreads();
    {
      uint4 da0 = *(const uint4*)(gA + k0);
      uint4 da1 = *(const uint4*)(gA + k0 + 8);
      uint4 db0 = *(const uint4*)(gB + k0);
      uint4 db1 = *(const uint4*)(gB + k0 + 8);
      *(uint4*)(&Asl[cur][srow * 40 + scol]) = da0;
      *(uint4*)(&Asl[cur][srow * 40 + scol + 8]) = da1;
      *(uint4*)(&Btl[cur][srow * 40 + scol]) = db0;
      *(uint4*)(&Btl[cur][srow * 40 + scol + 8]) = db1;
    }
    __syncthreads();
#endif

    v16h af[2];
#pragma unroll
    for (int mt = 0; mt < 2; ++mt) {
      const h16* p = &Asl[cur][(wm * 32 + mt * 16 + l16) * 40];
      v8h lo = *(const v8h*)(p + hl * 8);
      v8h hi = *(const v8h*)(p + 16 + hl * 8);
#pragma unroll
      for (int j = 0; j < 8; ++j) { af[mt][j] = lo[j]; af[mt][8 + j] = hi[j]; }
    }
    v16h bf[4];
#pragma unroll
    for (int nt = 0; nt < 4; ++nt) {
      const h16* p = &Btl[cur][(wn * 64 + nt * 16 + l16) * 40 + hl * 8];
      v8h lo = *(const v8h*)(p);
      v8h hi = *(const v8h*)(p + 16);
#pragma unroll
      for (int j = 0; j < 8; ++j) { bf[nt][j] = lo[j]; bf[nt][8 + j] = hi[j]; }
    }
#pragma unroll
    for (int mt = 0; mt < 2; ++mt)
#pragma unroll
      for (int nt = 0; nt < 4; ++nt)
        acc[mt][nt] = wmma16(af[mt], bf[nt], acc[mt][nt]);
  }

  // ----- epilogue
#pragma unroll
  for (int nt = 0; nt < 4; ++nt) {
    const int n = n0 + wn * 64 + nt * 16 + l16;
    const float bv = bias[n] * bscale;
#pragma unroll
    for (int mt = 0; mt < 2; ++mt) {
      const int mbase = m0 + wm * 32 + mt * 16 + hl * 8;
      if (OM == 2) {
        // V^T store: [B,H,DH,S]; per lane 8 consecutive s -> one v8h store
        const int b = mbase / SS, s = mbase % SS;
        const int hh = n >> 6, d = n & 63;
        v8h pk;
#pragma unroll
        for (int r = 0; r < 8; ++r) pk[r] = (h16)(acc[mt][nt][r] + bv);
        *(v8h*)((h16*)Cout + (((size_t)b * HH + hh) * DH + d) * SS + s) = pk;
      } else {
#pragma unroll
        for (int r = 0; r < 8; ++r) {
          const float v = acc[mt][nt][r] + bv;
          if (OM == 1) ((h16*)Cout)[(size_t)(mbase + r) * N + n] = (h16)v;
          else         ((float*)Cout)[(size_t)(mbase + r) * N + n] = v;
        }
      }
    }
  }
}

// ---------------------------------------------------------------- attention
// One workgroup per (b, h, 128-query block); 8 waves, each owns 16 q rows.
// Q is pre-scaled by log2(e)/sqrt(DH) (folded into Wq/bq), softmax in base 2.
__global__ __launch_bounds__(256)
void flash_attn(const h16* __restrict__ Q, const h16* __restrict__ Kx,
                const h16* __restrict__ Vtg, h16* __restrict__ AO) {
  __shared__ h16 Kl[2][64 * 72];     // [kv][d]
  __shared__ h16 Vt[2][64 * 72];     // [d][kv]
  __shared__ h16 Pl[8 * 16 * 72];    // per-wave P tile [16][64]

  const int tid  = threadIdx.x;
  const int wave = tid >> 5;
  const int lane = tid & 31;
  const int hl   = lane >> 4;
  const int l16  = lane & 15;
  const int qb = blockIdx.x, h = blockIdx.y, b = blockIdx.z;
  const int qrow0 = qb * 128 + wave * 16;
  const size_t baseQ = ((size_t)b * SS + qrow0) * DM + h * DH;

  // staging slots: row = tid>>2 (0..63), 32B chunk = (tid&3)*16 halves
  const int srow = tid >> 2;
  const int scol = (tid & 3) * 16;
  const h16* gK = Kx + ((size_t)b * SS + srow) * DM + h * DH + scol;
  const h16* gV = Vtg + (((size_t)b * HH + h) * DH + srow) * SS + scol;
  const int lofs = srow * 72 + scol;

  // Q fragments: 2 k-chunks of 32, in registers for the whole loop
  v16h qf[2];
#pragma unroll
  for (int c = 0; c < 2; ++c) {
    const h16* p = Q + baseQ + (size_t)l16 * DM + c * 32;
    v8h lo = *(const v8h*)(p + hl * 8);
    v8h hi = *(const v8h*)(p + 16 + hl * 8);
#pragma unroll
    for (int j = 0; j < 8; ++j) { qf[c][j] = lo[j]; qf[c][8 + j] = hi[j]; }
  }

  v8f o[4];
#pragma unroll
  for (int nt = 0; nt < 4; ++nt)
#pragma unroll
    for (int r = 0; r < 8; ++r) o[nt][r] = 0.f;
  float mrow[8], lrow[8];
#pragma unroll
  for (int r = 0; r < 8; ++r) { mrow[r] = -1e30f; lrow[r] = 0.f; }

  const int nkb = SS / 64;

#if USE_ASYNC
  {
    h16* lk = &Kl[0][lofs]; h16* lv = &Vt[0][lofs];
    CP16(gK, lk); CP16(gK + 8, lk + 8);
    CP16(gV, lv); CP16(gV + 8, lv + 8);
  }
#endif

  for (int kb = 0; kb < nkb; ++kb) {
    const int cur = kb & 1;
#if USE_ASYNC
    WAIT_ASYNC();
    __syncthreads();
    if (kb + 1 < nkb) {
      const size_t gk1 = (size_t)(kb + 1) * 64 * DM;  // K advances 64 rows
      const size_t gv1 = (size_t)(kb + 1) * 64;       // V^T advances 64 cols
      h16* lk = &Kl[1 - cur][lofs]; h16* lv = &Vt[1 - cur][lofs];
      CP16(gK + gk1, lk); CP16(gK + gk1 + 8, lk + 8);
      CP16(gV + gv1, lv); CP16(gV + gv1 + 8, lv + 8);
    }
#else
    __syncthreads();
    {
      uint4 dk0 = *(const uint4*)(gK + (size_t)kb * 64 * DM);
      uint4 dk1 = *(const uint4*)(gK + (size_t)kb * 64 * DM + 8);
      uint4 dv0 = *(const uint4*)(gV + (size_t)kb * 64);
      uint4 dv1 = *(const uint4*)(gV + (size_t)kb * 64 + 8);
      *(uint4*)(&Kl[cur][lofs]) = dk0;
      *(uint4*)(&Kl[cur][lofs + 8]) = dk1;
      *(uint4*)(&Vt[cur][lofs]) = dv0;
      *(uint4*)(&Vt[cur][lofs + 8]) = dv1;
    }
    __syncthreads();
#endif

    // S = Q @ K^T (Q pre-scaled)
    v8f sa[4];
#pragma unroll
    for (int nt = 0; nt < 4; ++nt) {
#pragma unroll
      for (int r = 0; r < 8; ++r) sa[nt][r] = 0.f;
#pragma unroll
      for (int c = 0; c < 2; ++c) {
        const h16* p = &Kl[cur][(nt * 16 + l16) * 72 + c * 32 + hl * 8];
        v8h lo = *(const v8h*)(p);
        v8h hi = *(const v8h*)(p + 16);
        v16h bf;
#pragma unroll
        for (int j = 0; j < 8; ++j) { bf[j] = lo[j]; bf[8 + j] = hi[j]; }
        sa[nt] = wmma16(qf[c], bf, sa[nt]);
      }
    }

    // online softmax, base-2 (rows M = hl*8 + r)
    float alpha[8];
#pragma unroll
    for (int r = 0; r < 8; ++r) {
      float mx = fmaxf(fmaxf(sa[0][r], sa[1][r]), fmaxf(sa[2][r], sa[3][r]));
      mx = redmax16(mx);
      const float mn = fmaxf(mrow[r], mx);
      alpha[r] = EXP2F(mrow[r] - mn);
      mrow[r] = mn;
    }
    float rs[8];
#pragma unroll
    for (int r = 0; r < 8; ++r) rs[r] = 0.f;
#pragma unroll
    for (int nt = 0; nt < 4; ++nt)
#pragma unroll
      for (int r = 0; r < 8; ++r) {
        const float p = EXP2F(sa[nt][r] - mrow[r]);
        sa[nt][r] = p;
        rs[r] += p;
      }
#pragma unroll
    for (int r = 0; r < 8; ++r) {
      lrow[r] = lrow[r] * alpha[r] + redsum16(rs[r]);
#pragma unroll
      for (int nt = 0; nt < 4; ++nt) o[nt][r] *= alpha[r];
    }

    // C-layout P -> LDS -> A-layout fragments (per-wave region, wave-order DS)
    h16* pw = &Pl[wave * 16 * 72];
#pragma unroll
    for (int nt = 0; nt < 4; ++nt)
#pragma unroll
      for (int r = 0; r < 8; ++r)
        pw[(hl * 8 + r) * 72 + nt * 16 + l16] = (h16)sa[nt][r];

    v16h pf[2];
#pragma unroll
    for (int c = 0; c < 2; ++c) {
      const h16* pp = &Pl[(wave * 16 + l16) * 72 + c * 32];
      v8h lo = *(const v8h*)(pp + hl * 8);
      v8h hi = *(const v8h*)(pp + 16 + hl * 8);
#pragma unroll
      for (int j = 0; j < 8; ++j) { pf[c][j] = lo[j]; pf[c][8 + j] = hi[j]; }
    }

    // O += P @ V
#pragma unroll
    for (int nt = 0; nt < 4; ++nt)
#pragma unroll
      for (int c = 0; c < 2; ++c) {
        const h16* p = &Vt[cur][(nt * 16 + l16) * 72 + c * 32 + hl * 8];
        v8h lo = *(const v8h*)(p);
        v8h hi = *(const v8h*)(p + 16);
        v16h bf;
#pragma unroll
        for (int j = 0; j < 8; ++j) { bf[j] = lo[j]; bf[8 + j] = hi[j]; }
        o[nt] = wmma16(pf[c], bf, o[nt]);
      }
  }

  // finalize + store f16 [B*S][DM]
#pragma unroll
  for (int r = 0; r < 8; ++r) {
    const float inv = 1.0f / lrow[r];
#pragma unroll
    for (int nt = 0; nt < 4; ++nt)
      AO[((size_t)b * SS + qrow0 + hl * 8 + r) * DM + h * DH + nt * 16 + l16] =
          (h16)(o[nt][r] * inv);
  }
}

// ---------------------------------------------------------------- launch
extern "C" void kernel_launch(void* const* d_in, const int* in_sizes, int n_in,
                              void* d_out, int out_size, void* d_ws, size_t ws_size,
                              hipStream_t stream) {
  const float* X1 = (const float*)d_in[0];
  const float* X2 = (const float*)d_in[1];
  const float* Wq = (const float*)d_in[2];
  const float* bq = (const float*)d_in[3];
  const float* Wk = (const float*)d_in[4];
  const float* bk = (const float*)d_in[5];
  const float* Wv = (const float*)d_in[6];
  const float* bv = (const float*)d_in[7];
  const float* Wo = (const float*)d_in[8];
  const float* bo = (const float*)d_in[9];

  h16* ws = (h16*)d_ws;
  size_t off = 0;
  const size_t NX = (size_t)MTOT * DIN;
  const size_t NW = (size_t)DIN * DM;
  const size_t NM = (size_t)MTOT * DM;
  h16* X1h = ws + off; off += NX;
  h16* X2h = ws + off; off += NX;
  h16* Wqt = ws + off; off += NW;
  h16* Wkt = ws + off; off += NW;
  h16* Wvt = ws + off; off += NW;
  h16* Wot = ws + off; off += NW;
  h16* Qh  = ws + off; off += NM;
  h16* Kh  = ws + off; off += NM;
  h16* Vtg = ws + off; off += NM;   // V^T [B,H,DH,S]
  h16* AOh = ws + off; off += NM;

  const float qs = 0.125f * 1.4426950408889634f;  // 1/sqrt(DH) * log2(e)

  cvt_f32_f16<<<(int)(NX / 1024), 256, 0, stream>>>(X1, X1h, (int)NX);
  cvt_f32_f16<<<(int)(NX / 1024), 256, 0, stream>>>(X2, X2h, (int)NX);

  dim3 gt(DIN / 32, DM / 32);
  cvt_tr<<<gt, 256, 0, stream>>>(Wq, Wqt, DIN, DM, qs);
  cvt_tr<<<gt, 256, 0, stream>>>(Wk, Wkt, DIN, DM, 1.0f);
  cvt_tr<<<gt, 256, 0, stream>>>(Wv, Wvt, DIN, DM, 1.0f);
  cvt_tr<<<dim3(DM / 32, DM / 32), 256, 0, stream>>>(Wo, Wot, DM, DM, 1.0f);

  dim3 gg(DM / 128, MTOT / 128);  // (8, 64)
  gemm_wmma<1><<<gg, 256, 0, stream>>>(X1h, Wqt, bq, qs,   Qh,  MTOT, DM, DIN);
  gemm_wmma<1><<<gg, 256, 0, stream>>>(X2h, Wkt, bk, 1.0f, Kh,  MTOT, DM, DIN);
  gemm_wmma<2><<<gg, 256, 0, stream>>>(X2h, Wvt, bv, 1.0f, Vtg, MTOT, DM, DIN);

  dim3 ga(SS / 128, HH, BB);      // (16, 16, 4)
  flash_attn<<<ga, 256, 0, stream>>>(Qh, Kh, Vtg, AOh);

  gemm_wmma<0><<<gg, 256, 0, stream>>>(AOh, Wot, bo, 1.0f, d_out, MTOT, DM, DM);
}
